// TriplanarDecoder_29532195127948
// MI455X (gfx1250) — compile-verified, matrix operand order checked by reading, not run on hardware
//
#include <hip/hip_runtime.h>

typedef __attribute__((ext_vector_type(16))) _Float16     v16h;
typedef __attribute__((ext_vector_type(8)))  float        v8f;
typedef __attribute__((ext_vector_type(4)))  _Float16     h4;
typedef __attribute__((ext_vector_type(4)))  unsigned int u32x4;
typedef _Float16 half_t;

#define L_CH   128
#define HH     64
#define WW     64
#define NQ     131072
#define PADF   0.1f
#define DIN    131          // 128 feat + 3 coords
#define K1PAD  160          // layer-1 K padded to 5 * 32
#define HID    512
#define QB     64           // queries per workgroup
#define KT1    5            // 160 / 32
#define KT2    16           // 512 / 32

// ---------------------------------------------------------------------------
// Prep kernel 1: plane features fp32 [3*128][64][64] -> f16 [3][64][64][128]
// ---------------------------------------------------------------------------
__global__ __launch_bounds__(256) void pack_feats_kernel(
    const float* __restrict__ src, half_t* __restrict__ dst) {
  int i = blockIdx.x * 256 + threadIdx.x;     // over 3*64*64*128 elements
  int c  = i & (L_CH - 1);
  int s  = i >> 7;                            // p*4096 + y*64 + x
  int p  = s >> 12;
  int yx = s & 4095;
  dst[i] = (half_t)src[((p * L_CH + c) << 12) + yx];
}

// ---------------------------------------------------------------------------
// Prep kernel 2: weights fp32 [Kreal][512] row-major -> f16 packed WMMA B
// fragments. Fragment f = nt*KT + kt holds the 32x16 tile (K = kt*32..,
// N = nt*16..). Per ISA 16-bit B layout: lanes 0-15 take K {0..7,16..23},
// lanes 16-31 take K {8..15,24..31}; each lane stores 16 halves (32B).
// ---------------------------------------------------------------------------
__global__ __launch_bounds__(256) void pack_weights_kernel(
    const float* __restrict__ W, half_t* __restrict__ dst, int Kreal, int KT) {
  int i    = blockIdx.x * 256 + threadIdx.x;  // frag*32 + lane
  int lane = i & 31;
  int f    = i >> 5;
  int kt   = f % KT;
  int nt   = f / KT;
  int n    = nt * 16 + (lane & 15);
  int base = (lane < 16) ? 0 : 8;
  half_t vals[16];
#pragma unroll
  for (int v = 0; v < 8; ++v) {
    int k0 = kt * 32 + base + 2 * v + ((v >= 4) ? 8 : 0);
#pragma unroll
    for (int j = 0; j < 2; ++j) {
      int k = k0 + j;
      vals[2 * v + j] = (k < Kreal) ? (half_t)W[k * HID + n] : (half_t)0.0f;
    }
  }
  *(v16h*)(dst + (size_t)i * 16) = *(v16h*)vals;
}

// ---------------------------------------------------------------------------
// One MLP layer as LDS-resident WMMA GEMM. Each wave iteration computes a
// 16(M) x 32(N) slab as two 16x16 WMMA tiles sharing one A fragment -> two
// independent accumulator chains feeding the XDL pipe. A fragments are two
// contiguous 16B runs per lane -> 2 x ds_load_b128 each. Loops are
// wave-uniform so EXEC is all-ones at every WMMA.
// ---------------------------------------------------------------------------
template <int KT>
__device__ __forceinline__ void gemm_layer(
    const half_t* __restrict__ Asrc, int Astride,
    const half_t* __restrict__ Wpack, const float* __restrict__ bias,
    half_t* __restrict__ Dst, int Dstride, int wave, int lane) {
  const int MT    = QB / 16;          // 4 row tiles
  const int NPAIR = (HID / 16) / 2;   // 16 column-tile pairs
  const int kbase = (lane < 16) ? 0 : 8;
  for (int t = wave; t < MT * NPAIR; t += 8) {
    const int mt  = t & (MT - 1);     // wave-invariant: A stays hoistable
    const int np  = t >> 2;
    const int nt0 = 2 * np;

    const float bv0 = bias[nt0 * 16 + (lane & 15)];
    const float bv1 = bias[nt0 * 16 + 16 + (lane & 15)];
    v8f acc0 = {bv0, bv0, bv0, bv0, bv0, bv0, bv0, bv0};
    v8f acc1 = {bv1, bv1, bv1, bv1, bv1, bv1, bv1, bv1};

    const half_t* __restrict__ arow = Asrc + (mt * 16 + (lane & 15)) * Astride;
    const v16h* __restrict__ bp0 =
        (const v16h*)Wpack + ((size_t)nt0 * KT) * 32 + lane;
    const v16h* __restrict__ bp1 = bp0 + (size_t)KT * 32;

#pragma unroll
    for (int kt = 0; kt < KT; ++kt) {
      v16h a;
      *(u32x4*)((unsigned int*)&a + 0) =
          *(const u32x4*)(arow + kt * 32 + kbase);        // ds_load_b128
      *(u32x4*)((unsigned int*)&a + 4) =
          *(const u32x4*)(arow + kt * 32 + kbase + 16);   // ds_load_b128
      v16h b0 = bp0[(size_t)kt * 32];
      v16h b1 = bp1[(size_t)kt * 32];
      acc0 = __builtin_amdgcn_wmma_f32_16x16x32_f16(
          false, a, false, b0, (short)0, acc0, false, false);
      acc1 = __builtin_amdgcn_wmma_f32_16x16x32_f16(
          false, a, false, b1, (short)0, acc1, false, false);
    }

    // C/D layout: VGPR r -> M = r (+8 for lanes 16-31), N = lane & 15
    const int drow0 = mt * 16 + ((lane < 16) ? 0 : 8);
    const int dcol0 = nt0 * 16 + (lane & 15);
#pragma unroll
    for (int r = 0; r < 8; ++r) {
      float x0 = acc0[r];
      float x1 = acc1[r];
      x0 = (x0 > 0.0f) ? x0 : 0.0f;   // ReLU
      x1 = (x1 > 0.0f) ? x1 : 0.0f;
      Dst[(drow0 + r) * Dstride + dcol0]      = (half_t)x0;
      Dst[(drow0 + r) * Dstride + dcol0 + 16] = (half_t)x1;
    }
  }
}

// ---------------------------------------------------------------------------
// Fused triplanar sample + 4-layer MLP. One workgroup = 64 queries.
// ---------------------------------------------------------------------------
__global__ __launch_bounds__(256) void triplanar_mlp_kernel(
    const half_t* __restrict__ featp,   // [3][64][64][128] f16
    const float*  __restrict__ query,   // [N][3]
    const half_t* __restrict__ w1p, const float* __restrict__ b1,
    const half_t* __restrict__ w2p, const float* __restrict__ b2,
    const half_t* __restrict__ w3p, const float* __restrict__ b3,
    const float*  __restrict__ Wo,  const float* __restrict__ bo,
    float* __restrict__ out) {
  extern __shared__ char smem[];
  half_t* bufA = (half_t*)smem;                                   // QB x K1PAD
  half_t* bufB = (half_t*)(smem + QB * K1PAD * 2);                // QB x HID
  half_t* bufC = (half_t*)(smem + QB * K1PAD * 2 + QB * HID * 2); // QB x HID

  const int tid   = threadIdx.x;
  const int wave  = tid >> 5;
  const int lane  = tid & 31;
  const int qbase = blockIdx.x * QB;
  const float inv = 1.0f / (1.0f + PADF);

  // ---- Phase 0: triplanar bilinear sampling into bufA (f16) --------------
  // Wave handles 8 queries; 32 lanes cover 128 channels (4 each).
  for (int qi = 0; qi < 8; ++qi) {
    const int q  = wave * 8 + qi;
    const int qg = qbase + q;
    const float qx = query[qg * 3 + 0];
    const float qy = query[qg * 3 + 1];
    const float qz = query[qg * 3 + 2];
    const int c = lane * 4;
    float a0 = 0.f, a1 = 0.f, a2 = 0.f, a3 = 0.f;

#pragma unroll
    for (int p = 0; p < 3; ++p) {
      // plane uv: p0=(x,z), p1=(y,z), p2=(x,y)
      float u = ((p == 1) ? qy : qx) * inv;
      float v = ((p == 2) ? qy : qz) * inv;
      float x = (u + 1.0f) * 0.5f * (float)(WW - 1);
      float y = (v + 1.0f) * 0.5f * (float)(HH - 1);
      float x0 = floorf(x), y0 = floorf(y);
      float wx = x - x0, wy = y - y0;
      int x0i = (int)x0;      if (x0i < 0) x0i = 0; if (x0i > WW - 1) x0i = WW - 1;
      int x1i = (int)x0 + 1;  if (x1i < 0) x1i = 0; if (x1i > WW - 1) x1i = WW - 1;
      int y0i = (int)y0;      if (y0i < 0) y0i = 0; if (y0i > HH - 1) y0i = HH - 1;
      int y1i = (int)y0 + 1;  if (y1i < 0) y1i = 0; if (y1i > HH - 1) y1i = HH - 1;

      const half_t* __restrict__ pb = featp + (size_t)p * HH * WW * L_CH;
      const float wgt[4] = {(1.f - wx) * (1.f - wy), wx * (1.f - wy),
                            (1.f - wx) * wy,          wx * wy};
      const int idx[4] = {y0i * WW + x0i, y0i * WW + x1i,
                          y1i * WW + x0i, y1i * WW + x1i};
#pragma unroll
      for (int t4 = 0; t4 < 4; ++t4) {
        h4 d = *(const h4*)(pb + (size_t)idx[t4] * L_CH + c);   // 8B coalesced
        a0 += wgt[t4] * (float)d[0];
        a1 += wgt[t4] * (float)d[1];
        a2 += wgt[t4] * (float)d[2];
        a3 += wgt[t4] * (float)d[3];
      }
    }
    half_t* arow = bufA + q * K1PAD;
    arow[c + 0] = (half_t)a0;
    arow[c + 1] = (half_t)a1;
    arow[c + 2] = (half_t)a2;
    arow[c + 3] = (half_t)a3;
    // zero K pad 128..159 then drop in the raw coords at 128..130
    if (lane < 16) *(unsigned int*)(arow + 128 + 2 * lane) = 0u;
    if (lane == 0) {
      arow[128] = (half_t)qx;
      arow[129] = (half_t)qy;
      arow[130] = (half_t)qz;
    }
  }
  __syncthreads();

  // ---- Phase 1-3: WMMA GEMM layers ---------------------------------------
  gemm_layer<KT1>(bufA, K1PAD, w1p, b1, bufB, HID, wave, lane);
  __syncthreads();
  gemm_layer<KT2>(bufB, HID,   w2p, b2, bufC, HID, wave, lane);
  __syncthreads();
  gemm_layer<KT2>(bufC, HID,   w3p, b3, bufB, HID, wave, lane);
  __syncthreads();

  // ---- Phase 4: 512 -> 1 head + tanh, all 256 threads --------------------
  // 4 lanes per query, each sums a 128-wide segment, shuffle-reduce.
  {
    const int q   = tid >> 2;   // 0..63
    const int seg = tid & 3;    // 0..3
    const half_t* __restrict__ h = bufB + q * HID + seg * 128;
    const float*  __restrict__ w = Wo + seg * 128;
    float s = 0.0f;
#pragma unroll 8
    for (int k = 0; k < 128; k += 4) {
      h4 hv = *(const h4*)(h + k);
      float4 wv = *(const float4*)(w + k);
      s = fmaf((float)hv[0], wv.x, s);
      s = fmaf((float)hv[1], wv.y, s);
      s = fmaf((float)hv[2], wv.z, s);
      s = fmaf((float)hv[3], wv.w, s);
    }
    s += __shfl_xor(s, 1);
    s += __shfl_xor(s, 2);
    if (seg == 0) out[qbase + q] = tanhf(s + bo[0]);
  }
}

// ---------------------------------------------------------------------------
// Launch. Inputs: plane_features, query, W1,b1, W2,b2, W3,b3, Wo,bo
// Workspace layout (bytes, 256-aligned offsets):
//   featp : 3*64*64*128 f16          = 3,145,728
//   w1p   : 32*5 *32*16 f16 (frags)  =   163,840
//   w2p   : 32*16*32*16 f16          =   524,288
//   w3p   : 32*16*32*16 f16          =   524,288
// ---------------------------------------------------------------------------
extern "C" void kernel_launch(void* const* d_in, const int* in_sizes, int n_in,
                              void* d_out, int out_size, void* d_ws, size_t ws_size,
                              hipStream_t stream) {
  const float* planes = (const float*)d_in[0];
  const float* query  = (const float*)d_in[1];
  const float* W1 = (const float*)d_in[2];
  const float* b1 = (const float*)d_in[3];
  const float* W2 = (const float*)d_in[4];
  const float* b2 = (const float*)d_in[5];
  const float* W3 = (const float*)d_in[6];
  const float* b3 = (const float*)d_in[7];
  const float* Wo = (const float*)d_in[8];
  const float* bo = (const float*)d_in[9];
  float* out = (float*)d_out;

  char* ws = (char*)d_ws;
  half_t* featp = (half_t*)(ws);
  half_t* w1p   = (half_t*)(ws + 3145728);
  half_t* w2p   = (half_t*)(ws + 3145728 + 163840);
  half_t* w3p   = (half_t*)(ws + 3145728 + 163840 + 524288);

  // Prep: feature repack (channels-last f16) + WMMA-fragment weight packs
  pack_feats_kernel<<<(3 * HH * WW * L_CH) / 256, 256, 0, stream>>>(planes, featp);
  pack_weights_kernel<<<(32 * KT1 * 32) / 256, 256, 0, stream>>>(W1, w1p, DIN, KT1);
  pack_weights_kernel<<<(32 * KT2 * 32) / 256, 256, 0, stream>>>(W2, w2p, HID, KT2);
  pack_weights_kernel<<<(32 * KT2 * 32) / 256, 256, 0, stream>>>(W3, w3p, HID, KT2);

  // Fused sample + MLP
  const size_t lds_bytes = (size_t)(QB * K1PAD + 2 * QB * HID) * sizeof(half_t); // 148 KB
  triplanar_mlp_kernel<<<NQ / QB, 256, lds_bytes, stream>>>(
      featp, query, w1p, b1, w2p, b2, w3p, b3, Wo, bo, out);
}
